// ImplicitFourierVolume_73770358276377
// MI455X (gfx1250) — compile-verified, matrix operand order checked by reading, not run on hardware
//
#include <hip/hip_runtime.h>
#include <hip/hip_bf16.h>

typedef float    v2f  __attribute__((ext_vector_type(2)));
typedef float    v4f  __attribute__((ext_vector_type(4)));
typedef float    v8f  __attribute__((ext_vector_type(8)));
typedef _Float16 v8h  __attribute__((ext_vector_type(8)));
typedef _Float16 v16h __attribute__((ext_vector_type(16)));

#define ZDIM    8
#define PEDIM   64
#define INDIM   136            // ZDIM + 2*PEDIM
#define KPAD0   160            // INDIM zero-padded to a multiple of 32
#define HIDDEN  128
#define NTILES  8              // HIDDEN / 16
#define LSTRIDE 164            // f32 LDS row stride: 16B-aligned, 164%64=36, 36/4=9 odd
                               //  -> conflict-free b128 access across 16 rows

// f16 transposed-weight workspace layout (offsets in _Float16 elements)
#define W0T_OFF   0                        // [128][160]
#define W1T_OFF   (W0T_OFF + 128 * KPAD0)  // [128][128]
#define W2T_OFF   (W1T_OFF + 128 * 128)    // [128][128]
#define WOUTT_OFF (W2T_OFF + 128 * 128)    // [16][128]
#define WS_HALVES (WOUTT_OFF + 16 * 128)
#define WS_NEED_BYTES ((size_t)WS_HALVES * 2)

// ---------------------------------------------------------------------------
// Zero-fill the output grid (harness poisons d_out; unmasked pixels must be 0)
// ---------------------------------------------------------------------------
__global__ void ifv_zero_kernel(float* __restrict__ out, int n) {
    int i = blockIdx.x * blockDim.x + threadIdx.x;
    if (i < n) out[i] = 0.0f;
}

// ---------------------------------------------------------------------------
// Prologue: transpose + convert f32 weights [K][C] -> f16 [Cpad][Kpad],
// zero-padding rows/cols so GEMM K-loops can run in 32-wide steps.
// ---------------------------------------------------------------------------
__global__ void ifv_convert_wT(const float* __restrict__ src, _Float16* __restrict__ dst,
                               int K, int C, int Kpad, int Cpad) {
    int idx = blockIdx.x * blockDim.x + threadIdx.x;
    int total = Cpad * Kpad;
    if (idx >= total) return;
    int c = idx / Kpad, k = idx - c * Kpad;
    float v = (k < K && c < C) ? src[k * C + c] : 0.0f;
    dst[idx] = (_Float16)v;
}

// ---------------------------------------------------------------------------
// Shared feature stage: build the 16 x KPAD0 f32 feature tile in LDS.
// cols [0,8) = z, [8,72) = sin, [72,136) = cos, [136,160) = 0 (K padding).
// ---------------------------------------------------------------------------
__device__ __forceinline__ void build_features(
    float* buf, const float* __restrict__ z, const float* __restrict__ rotmat,
    const float* __restrict__ pcoord, const int* __restrict__ mask_idx,
    const float* __restrict__ bg, int tile, int M, int R, int lane) {
    const int rrow = lane & 15;
    const int half = lane >> 4;

    int r = tile * 16 + rrow;
    if (r >= R) r = R - 1;                         // padded rows: duplicate
    const int b = r / M;
    const int m = r - b * M;
    const int pix = mask_idx[m];
    const float p0 = pcoord[pix * 3 + 0];
    const float p1 = pcoord[pix * 3 + 1];
    const float p2 = pcoord[pix * 3 + 2];
    const float* rm = rotmat + b * 9;
    const float s0 = p0 * rm[0] + p1 * rm[3] + p2 * rm[6];
    const float s1 = p0 * rm[1] + p1 * rm[4] + p2 * rm[7];
    const float s2 = p0 * rm[2] + p1 * rm[5] + p2 * rm[8];

    if (half == 0) {
#pragma unroll
        for (int i = 0; i < ZDIM; ++i)
            buf[rrow * LSTRIDE + i] = z[b * ZDIM + i];
    }
    const float twopi = 6.28318530717958647692f;
#pragma unroll 4
    for (int j = 0; j < 32; ++j) {
        const int p = half * 32 + j;               // lane pair splits 64 freqs
        const float pr = twopi * (s0 * bg[p] + s1 * bg[64 + p] + s2 * bg[128 + p]);
        buf[rrow * LSTRIDE + ZDIM + p]         = __sinf(pr);
        buf[rrow * LSTRIDE + ZDIM + PEDIM + p] = __cosf(pr);
    }
    // zero K-padding columns [INDIM, KPAD0)
    for (int i = lane; i < 16 * (KPAD0 - INDIM); i += 32) {
        int rr = i / (KPAD0 - INDIM), cc = i - rr * (KPAD0 - INDIM);
        buf[rr * LSTRIDE + INDIM + cc] = 0.0f;
    }
    asm volatile("s_wait_dscnt 0x0" ::: "memory");
}

// ===========================================================================
// PRIMARY PATH: f16 WMMA (V_WMMA_F32_16X16X32_F16), f32 accumulate.
// ===========================================================================

// A fragment (ISA 7.12.2, 16-bit A 16x32): per lane, elements 0-7 hold
// K = k0 + 8*half + 0..7 ; elements 8-15 hold K = k0 + 16 + 8*half + 0..7.
// Sourced from the f32 LDS tile with on-the-fly f32->f16 conversion.
__device__ __forceinline__ v16h load_a_frag16(const float* buf, int rrow, int half, int k0) {
    const v4f* ap = (const v4f*)(buf + rrow * LSTRIDE + k0 + 8 * half);
    v4f a0 = ap[0], a1 = ap[1];                    // K offsets 0..7
    v4f a2 = ap[4], a3 = ap[5];                    // K offsets 16..23
    v16h a;
#pragma unroll
    for (int i = 0; i < 4; ++i) {
        a[i]      = (_Float16)a0[i];
        a[4 + i]  = (_Float16)a1[i];
        a[8 + i]  = (_Float16)a2[i];
        a[12 + i] = (_Float16)a3[i];
    }
    return a;
}

// B fragment mirrors A's K striping with N = lane & 15; wrow points at
// Wt[col][k0 + 8*half], rows of Wt are Kpad halves (16B aligned).
__device__ __forceinline__ v16h load_b_frag16(const _Float16* __restrict__ wrow) {
    const v8h* bp = (const v8h*)wrow;
    v8h lo = bp[0];                                // K offsets 0..7
    v8h hi = bp[2];                                // K offsets 16..23
    v16h b;
#pragma unroll
    for (int i = 0; i < 8; ++i) { b[i] = lo[i]; b[8 + i] = hi[i]; }
    return b;
}

template <int KPAD>
__device__ __forceinline__ void gemm16_relu_layer(float* buf,
                                                  const _Float16* __restrict__ Wt,
                                                  const float* __restrict__ Bv,
                                                  int rrow, int half) {
    v8f acc[NTILES];
#pragma unroll
    for (int t = 0; t < NTILES; ++t) {
        float bv = Bv[t * 16 + rrow];              // bias depends on N only
        v8f c = {bv, bv, bv, bv, bv, bv, bv, bv};
        acc[t] = c;
    }
#pragma unroll
    for (int k0 = 0; k0 < KPAD; k0 += 32) {
        v16h a = load_a_frag16(buf, rrow, half, k0);
#pragma unroll
        for (int t = 0; t < NTILES; ++t) {
            v16h b = load_b_frag16(Wt + (size_t)(t * 16 + rrow) * KPAD + k0 + 8 * half);
            acc[t] = __builtin_amdgcn_wmma_f32_16x16x32_f16(
                false, a, false, b, (short)0, acc[t], false, false);
        }
    }
    // C/D layout: VGPR j holds M = j + 8*half, N = rrow. relu + f32 writeback.
#pragma unroll
    for (int t = 0; t < NTILES; ++t) {
#pragma unroll
        for (int j = 0; j < 8; ++j) {
            float v = acc[t][j];
            v = v > 0.0f ? v : 0.0f;
            buf[(j + 8 * half) * LSTRIDE + t * 16 + rrow] = v;
        }
    }
    asm volatile("s_wait_dscnt 0x0" ::: "memory");
}

__global__ __launch_bounds__(128) void ifv_mlp_f16_kernel(
    const float* __restrict__ z,      const float* __restrict__ rotmat,
    const float* __restrict__ pcoord, const int*   __restrict__ mask_idx,
    const float* __restrict__ bg,
    const _Float16* __restrict__ wst,                     // f16 weight workspace
    const float* __restrict__ b0, const float* __restrict__ b1,
    const float* __restrict__ b2, const float* __restrict__ bout,
    float* __restrict__ out, int M, int R, int Npix, int n_tiles) {
    __shared__ __align__(16) float lds[4][16 * LSTRIDE];

    const int lane = threadIdx.x & 31;
    const int wave = threadIdx.x >> 5;
    const int tile = blockIdx.x * 4 + wave;
    if (tile >= n_tiles) return;                   // wave-uniform exit

    float* buf = lds[wave];
    const int rrow = lane & 15;
    const int half = lane >> 4;

    build_features(buf, z, rotmat, pcoord, mask_idx, bg, tile, M, R, lane);

    gemm16_relu_layer<KPAD0>(buf, wst + W0T_OFF, b0, rrow, half);
    gemm16_relu_layer<HIDDEN>(buf, wst + W1T_OFF, b1, rrow, half);
    gemm16_relu_layer<HIDDEN>(buf, wst + W2T_OFF, b2, rrow, half);

    // ---- 128 -> 2 head as one padded 16-col WMMA tile ---------------------
    {
        float bv = (rrow < 2) ? bout[rrow] : 0.0f;
        v8f hacc = {bv, bv, bv, bv, bv, bv, bv, bv};
#pragma unroll
        for (int k0 = 0; k0 < HIDDEN; k0 += 32) {
            v16h a = load_a_frag16(buf, rrow, half, k0);
            v16h b = load_b_frag16(wst + WOUTT_OFF + (size_t)rrow * HIDDEN + k0 + 8 * half);
            hacc = __builtin_amdgcn_wmma_f32_16x16x32_f16(
                false, a, false, b, (short)0, hacc, false, false);
        }
        // redistribute C frag through LDS so each lane owns one (row, out) pair
#pragma unroll
        for (int j = 0; j < 8; ++j)
            buf[(j + 8 * half) * LSTRIDE + rrow] = hacc[j];
        asm volatile("s_wait_dscnt 0x0" ::: "memory");

        const int r = tile * 16 + rrow;
        const int o = half;
        float val = buf[rrow * LSTRIDE + o];
        if (r < R) {
            const int b = r / M;
            const int m = r - b * M;
            const int pix = mask_idx[m];
            out[((long)b * Npix + pix) * 2 + o] = val;
        }
    }
}

// ===========================================================================
// FALLBACK PATH: exact fp32 WMMA (V_WMMA_F32_16X16X4_F32) — used when the
// workspace is too small to hold the f16 transposed weights.
// ===========================================================================
template <int K>
__device__ __forceinline__ void gemm32_relu_layer(float* buf,
                                                   const float* __restrict__ W,
                                                   const float* __restrict__ Bv,
                                                   int rrow, int half) {
    v8f acc[NTILES];
#pragma unroll
    for (int t = 0; t < NTILES; ++t) {
        float bv = Bv[t * 16 + rrow];
        v8f c = {bv, bv, bv, bv, bv, bv, bv, bv};
        acc[t] = c;
    }
    const int koff = 2 * half;
#pragma unroll 2
    for (int k0 = 0; k0 < K; k0 += 4) {
        const int ka = k0 + koff;
        v2f a;
        a.x = buf[rrow * LSTRIDE + ka];
        a.y = buf[rrow * LSTRIDE + ka + 1];
#pragma unroll
        for (int t = 0; t < NTILES; ++t) {
            const int col = t * 16 + rrow;
            v2f b;
            b.x = W[ka * HIDDEN + col];
            b.y = W[(ka + 1) * HIDDEN + col];
            acc[t] = __builtin_amdgcn_wmma_f32_16x16x4_f32(
                false, a, false, b, (short)0, acc[t], false, false);
        }
    }
#pragma unroll
    for (int t = 0; t < NTILES; ++t) {
#pragma unroll
        for (int j = 0; j < 8; ++j) {
            float v = acc[t][j];
            v = v > 0.0f ? v : 0.0f;
            buf[(j + 8 * half) * LSTRIDE + t * 16 + rrow] = v;
        }
    }
    asm volatile("s_wait_dscnt 0x0" ::: "memory");
}

__global__ __launch_bounds__(128) void ifv_mlp_f32_kernel(
    const float* __restrict__ z,      const float* __restrict__ rotmat,
    const float* __restrict__ pcoord, const int*   __restrict__ mask_idx,
    const float* __restrict__ bg,
    const float* __restrict__ w0, const float* __restrict__ b0,
    const float* __restrict__ w1, const float* __restrict__ b1,
    const float* __restrict__ w2, const float* __restrict__ b2,
    const float* __restrict__ wout, const float* __restrict__ bout,
    float* __restrict__ out, int M, int R, int Npix, int n_tiles) {
    __shared__ __align__(16) float lds[4][16 * LSTRIDE];

    const int lane = threadIdx.x & 31;
    const int wave = threadIdx.x >> 5;
    const int tile = blockIdx.x * 4 + wave;
    if (tile >= n_tiles) return;

    float* buf = lds[wave];
    const int rrow = lane & 15;
    const int half = lane >> 4;

    build_features(buf, z, rotmat, pcoord, mask_idx, bg, tile, M, R, lane);

    gemm32_relu_layer<INDIM>(buf, w0, b0, rrow, half);
    gemm32_relu_layer<HIDDEN>(buf, w1, b1, rrow, half);
    gemm32_relu_layer<HIDDEN>(buf, w2, b2, rrow, half);

    {
        const int r = tile * 16 + rrow;
        const int o = half;
        float acc = bout[o];
#pragma unroll 8
        for (int k = 0; k < HIDDEN; ++k)
            acc += buf[rrow * LSTRIDE + k] * wout[k * 2 + o];
        if (r < R) {
            const int b = r / M;
            const int m = r - b * M;
            const int pix = mask_idx[m];
            out[((long)b * Npix + pix) * 2 + o] = acc;
        }
    }
}

extern "C" void kernel_launch(void* const* d_in, const int* in_sizes, int n_in,
                              void* d_out, int out_size, void* d_ws, size_t ws_size,
                              hipStream_t stream) {
    const float* z        = (const float*)d_in[0];
    const float* rotmat   = (const float*)d_in[1];
    const float* pcoord   = (const float*)d_in[2];
    const int*   mask_idx = (const int*)  d_in[3];
    const float* bg       = (const float*)d_in[4];
    const float* w0       = (const float*)d_in[5];
    const float* b0       = (const float*)d_in[6];
    const float* w1       = (const float*)d_in[7];
    const float* b1       = (const float*)d_in[8];
    const float* w2       = (const float*)d_in[9];
    const float* b2       = (const float*)d_in[10];
    const float* wout     = (const float*)d_in[11];
    const float* bout     = (const float*)d_in[12];
    float* out = (float*)d_out;

    const int M    = in_sizes[3];              // masked pixel count
    const int B    = in_sizes[0] / ZDIM;       // batch
    const int Npix = in_sizes[2] / 3;          // 256*256
    const int R    = B * M;                    // total MLP rows
    const int n_tiles = (R + 15) / 16;
    const int blocks  = (n_tiles + 3) / 4;     // 4 waves per block

    ifv_zero_kernel<<<(out_size + 255) / 256, 256, 0, stream>>>(out, out_size);

    if (ws_size >= WS_NEED_BYTES) {
        _Float16* wst = (_Float16*)d_ws;
        ifv_convert_wT<<<(128 * KPAD0 + 255) / 256, 256, 0, stream>>>(
            w0, wst + W0T_OFF, INDIM, HIDDEN, KPAD0, HIDDEN);
        ifv_convert_wT<<<(128 * 128 + 255) / 256, 256, 0, stream>>>(
            w1, wst + W1T_OFF, HIDDEN, HIDDEN, HIDDEN, HIDDEN);
        ifv_convert_wT<<<(128 * 128 + 255) / 256, 256, 0, stream>>>(
            w2, wst + W2T_OFF, HIDDEN, HIDDEN, HIDDEN, HIDDEN);
        ifv_convert_wT<<<(16 * 128 + 255) / 256, 256, 0, stream>>>(
            wout, wst + WOUTT_OFF, HIDDEN, 2, HIDDEN, 16);
        ifv_mlp_f16_kernel<<<blocks, 128, 0, stream>>>(
            z, rotmat, pcoord, mask_idx, bg, wst, b0, b1, b2, bout,
            out, M, R, Npix, n_tiles);
    } else {
        ifv_mlp_f32_kernel<<<blocks, 128, 0, stream>>>(
            z, rotmat, pcoord, mask_idx, bg, w0, b0, w1, b1, w2, b2, wout, bout,
            out, M, R, Npix, n_tiles);
    }
}
